// MultiscaleMoEAMIL_58179626992116
// MI455X (gfx1250) — matrix operand order repaired and checked by reading, not compiled, not run on hardware
//
#include <hip/hip_runtime.h>
#include <stdint.h>

// ---------------- problem constants ----------------
#define NN    16384
#define CC    4
#define D20c  1536
#define D5c   1536
#define Dtot  3072
#define Hc    256
#define Ec    8
#define POSc  2048
#define FFc   128
#define GK    128   // gate hidden

// ---------------- d_out layout (floats) ----------------
#define OUT_LOGITS  0
#define OUT_PROB    4
#define OUT_YHAT    8
#define OUT_A       12
#define OUT_REFINED (12 + CC*NN)                 // 65548
#define OUT_M       (OUT_REFINED + CC*Hc)        // 66572
#define OUT_GATE    (OUT_M + CC*Hc)              // 67596

// ---------------- workspace layout (bytes) ----------------
constexpr size_t WS_PE20   = 0;                               // N*2048 bf16
constexpr size_t WS_PE5    = WS_PE20 + (size_t)NN*POSc*2;
constexpr size_t WS_XB20   = WS_PE5  + (size_t)NN*POSc*2;     // N*1536 bf16
constexpr size_t WS_XB5    = WS_XB20 + (size_t)NN*D20c*2;
constexpr size_t WS_GHB    = WS_XB5  + (size_t)NN*D5c*2;      // C*N*128 bf16
constexpr size_t WS_FUSED  = WS_GHB  + (size_t)CC*NN*GK*2;    // C*N*256 f32
constexpr size_t WS_FUSEDB = WS_FUSED+ (size_t)CC*NN*Hc*4;    // C*N*256 bf16
constexpr size_t WS_AH     = WS_FUSEDB+(size_t)CC*NN*Hc*2;    // C*N*128 f32
// transposed bf16 weights: Wt[n][k] (column-major of original [k][n])
constexpr size_t WS_WP20B  = WS_AH   + (size_t)CC*NN*128*4;   // [1536][2048]
constexpr size_t WS_WP5B   = WS_WP20B+ (size_t)POSc*D20c*2;
constexpr size_t WS_GW1B   = WS_WP5B + (size_t)POSc*D5c*2;    // 4 x [128][3072]
constexpr size_t WS_E20W1B = WS_GW1B + (size_t)CC*Dtot*GK*2;  // 16 x [256][1536]
constexpr size_t WS_E5W1B  = WS_E20W1B+(size_t)16*D20c*Hc*2;
constexpr size_t WS_E20W2B = WS_E5W1B+ (size_t)16*D5c*Hc*2;   // 16 x [256][256]
constexpr size_t WS_E5W2B  = WS_E20W2B+(size_t)16*Hc*Hc*2;
constexpr size_t WS_AW1B   = WS_E5W2B+ (size_t)16*Hc*Hc*2;    // 4 x [128][256]
constexpr size_t WS_ASTATS = WS_AW1B + (size_t)CC*Hc*128*2;   // 8 floats
// pre-LN projection Y (bf16, N x 1536 per scale) aliases FUSED+FUSEDB
// (Y dead before experts_kernel writes fused) -- exact fit: 2*50.3MB = 100.6MB
constexpr size_t WS_Y20B   = WS_FUSED;
constexpr size_t WS_Y5B    = WS_FUSED + (size_t)NN*D20c*2;

// ---------------- WMMA helpers ----------------
typedef __attribute__((ext_vector_type(16))) __bf16 v16bf;
typedef __attribute__((ext_vector_type(8)))  __bf16 v8bf;
typedef __attribute__((ext_vector_type(8)))  float  v8f;

union V16U { v16bf v; v8bf h[2]; };

__device__ inline v8f vzero8() {
  v8f v;
#pragma unroll
  for (int i = 0; i < 8; ++i) v[i] = 0.0f;
  return v;
}

// A-fragment: 16 rows x 32 K, bf16, row-major source (LDS).
// lanes0-15 row=lane, K={0..15}; lanes16-31 row=lane-16, K={16..31}
__device__ inline v16bf load_a_frag(const __bf16* base, int stride, int k0, int lane) {
  int m  = lane & 15;
  int kh = lane >> 4;
  const __bf16* p = base + (size_t)m * stride + k0 + 16 * kh;
  V16U u;
  u.h[0] = *(const v8bf*)p;
  u.h[1] = *(const v8bf*)(p + 8);
  return u.v;
}

// B-fragment: 32 K x 16 cols from TRANSPOSED weight Wt[n][k] (leading dim K).
__device__ inline v16bf load_b_frag(const __bf16* Wt, int K, int k0, int col0, int lane) {
  int n  = lane & 15;
  int kh = lane >> 4;
  return *(const v16bf*)(Wt + (size_t)(col0 + n) * K + k0 + 16 * kh);
}

__device__ inline v8f wmma_bf16(v16bf a, v16bf b, v8f c) {
  return __builtin_amdgcn_wmma_f32_16x16x32_bf16(false, a, false, b, (short)0, c, false, false);
}

__device__ inline float gelu_exact(float x) {
  return 0.5f * x * (1.0f + erff(x * 0.70710678118654752f));
}

// ---------------- kernels ----------------

// fp32 [K][Ncol] -> bf16 [Ncol][K], LDS-tiled 32x32, batch in blockIdx.z
__global__ void cvt_t_kernel(const float* __restrict__ src, __bf16* __restrict__ dst,
                             int K, int Ncol) {
  __shared__ __bf16 tile[32][33];
  size_t base = (size_t)blockIdx.z * K * Ncol;
  int k0 = blockIdx.x * 32;
  int n0 = blockIdx.y * 32;
  int tx = threadIdx.x & 31, ty = threadIdx.x >> 5;   // 32 x 8
#pragma unroll
  for (int r = ty; r < 32; r += 8)
    tile[r][tx] = (__bf16)src[base + (size_t)(k0 + r) * Ncol + n0 + tx];
  __syncthreads();
#pragma unroll
  for (int r = ty; r < 32; r += 8)
    dst[base + (size_t)(n0 + r) * K + k0 + tx] = tile[tx][r];
}

__global__ void posenc_kernel(const float* __restrict__ coords, __bf16* __restrict__ pe) {
  int idx = blockIdx.x * 256 + threadIdx.x;        // [0, N*2048)
  int n = idx >> 11;
  int f = idx & 2047;
  int which = f >> 10;
  int ff = f & 1023;
  int jj = ff >> 1;
  float ee = (float)(jj >> 1) * (1.0f / 256.0f);
  float invdim = __expf(-9.210340371976184f * ee); // 10000^-ee
  float arg = coords[n * 2 + which] * invdim;
  float v = (ff & 1) ? __cosf(arg) : __sinf(arg);
  pe[idx] = (__bf16)v;
}

// tiled GEMM: Y = pe @ Wp + bp, bf16 out.  grid (N/64, 1536/256), block 256.
// Each wave: 2 col-tiles x 4 row-tiles, B fragment reused 4x from registers.
__global__ void proj_gemm_kernel(const __bf16* __restrict__ pe, const __bf16* __restrict__ WpT,
                                 const float* __restrict__ bp, __bf16* __restrict__ Ybf) {
  extern __shared__ char smem[];
  __bf16* As = (__bf16*)smem;                    // 64 x 2048 (262144 B)
  int tid = threadIdx.x, lane = tid & 31, w = tid >> 5;
  int n0 = blockIdx.x * 64;
  int cb = blockIdx.y * 256;

  {
    const uint4* s = (const uint4*)(pe + (size_t)n0 * POSc);
    uint4* d = (uint4*)As;
    for (int q = tid; q < 64 * POSc / 8; q += 256) d[q] = s[q];
  }
  __syncthreads();

  int cn = lane & 15, kh = lane >> 4;
  v8f acc[4][2];
#pragma unroll
  for (int rt = 0; rt < 4; ++rt)
#pragma unroll
    for (int t = 0; t < 2; ++t) acc[rt][t] = vzero8();

  for (int kk = 0; kk < POSc / 32; ++kk) {
    v16bf b0 = load_b_frag(WpT, POSc, kk * 32, cb + (w * 2 + 0) * 16, lane);
    v16bf b1 = load_b_frag(WpT, POSc, kk * 32, cb + (w * 2 + 1) * 16, lane);
#pragma unroll
    for (int rt = 0; rt < 4; ++rt) {
      v16bf a = load_a_frag(As + (size_t)rt * 16 * POSc, POSc, kk * 32, lane);
      acc[rt][0] = wmma_bf16(a, b0, acc[rt][0]);
      acc[rt][1] = wmma_bf16(a, b1, acc[rt][1]);
    }
  }
#pragma unroll
  for (int rt = 0; rt < 4; ++rt)
#pragma unroll
    for (int t = 0; t < 2; ++t) {
      int col = cb + (w * 2 + t) * 16 + cn;
#pragma unroll
      for (int r = 0; r < 8; ++r)
        Ybf[(size_t)(n0 + rt * 16 + r + 8 * kh) * D20c + col] = (__bf16)(acc[rt][t][r] + bp[col]);
    }
}

// LN + residual: xb = bf16( x[:,off:] + LN(Y) ),  grid N/16, block 256
__global__ void ln_kernel(const __bf16* __restrict__ Ybf, const float* __restrict__ gln,
                          const float* __restrict__ bln, const float* __restrict__ x,
                          int xoff, __bf16* __restrict__ xb) {
  extern __shared__ char smem[];
  __bf16* Ys   = (__bf16*)smem;                       // 16*1536 (49152 B)
  float*  redS = (float*)(smem + 49152);              // 256
  float*  redQ = (float*)(smem + 50176);              // 256
  float*  mu   = (float*)(smem + 51200);              // 16
  float*  rsg  = (float*)(smem + 51264);              // 16
  int tid = threadIdx.x;
  int n0 = blockIdx.x * 16;

  {
    const uint4* s = (const uint4*)(Ybf + (size_t)n0 * D20c);
    uint4* d = (uint4*)Ys;
    for (int q = tid; q < 16 * D20c / 8; q += 256) d[q] = s[q];
  }
  __syncthreads();
  {
    int rr = tid >> 4, tt = tid & 15;
    float s = 0.f, s2 = 0.f;
    for (int c = tt; c < D20c; c += 16) { float v = (float)Ys[rr * D20c + c]; s += v; s2 += v * v; }
    redS[tid] = s; redQ[tid] = s2;
  }
  __syncthreads();
  if (tid < 16) {
    float s = 0.f, s2 = 0.f;
    for (int t = 0; t < 16; ++t) { s += redS[tid * 16 + t]; s2 += redQ[tid * 16 + t]; }
    float m = s * (1.0f / D20c);
    float var = s2 * (1.0f / D20c) - m * m;
    mu[tid] = m; rsg[tid] = rsqrtf(var + 1e-5f);
  }
  __syncthreads();
  for (int q = 0; q < 96; ++q) {
    int e = q * 256 + tid;
    int row = e / D20c, col = e - row * D20c;
    float yn = ((float)Ys[e] - mu[row]) * rsg[row] * gln[col] + bln[col];
    float out = x[(size_t)(n0 + row) * Dtot + xoff + col] + yn;
    xb[(size_t)n0 * D20c + e] = (__bf16)out;
  }
}

// gate hidden: 64-row blocks, K split into x20 phase then x5 phase.
__global__ void gate1_kernel(const __bf16* __restrict__ xb20, const __bf16* __restrict__ xb5,
                             const __bf16* __restrict__ gW1T, const float* __restrict__ gb1,
                             __bf16* __restrict__ ghb) {
  extern __shared__ char smem[];
  __bf16* xs = (__bf16*)smem;   // 64 x 1536 (196608 B), reused per scale
  int tid = threadIdx.x, lane = tid & 31, w = tid >> 5;
  int n0 = blockIdx.x * 64;
  int cn = lane & 15, kh = lane >> 4;
  int col0 = w * 16;            // 8 waves x 16 = 128 cols

  v8f acc[CC][4];
#pragma unroll
  for (int c = 0; c < CC; ++c)
#pragma unroll
    for (int rt = 0; rt < 4; ++rt) acc[c][rt] = vzero8();

  // phase 0: x20 (K 0..1535), phase 1: x5 (K 1536..3071)
  for (int ph = 0; ph < 2; ++ph) {
    const __bf16* src = ph ? xb5 : xb20;
    {
      const uint4* s = (const uint4*)(src + (size_t)n0 * D20c);
      uint4* d = (uint4*)xs;
      for (int q = tid; q < 64 * D20c / 8; q += 256) d[q] = s[q];
    }
    __syncthreads();
#pragma unroll
    for (int c = 0; c < CC; ++c) {
      const __bf16* W = gW1T + (size_t)c * Dtot * GK;      // [128][3072]
      for (int kk = 0; kk < D20c / 32; ++kk) {
        v16bf b = load_b_frag(W, Dtot, ph * D20c + kk * 32, col0, lane);
#pragma unroll
        for (int rt = 0; rt < 4; ++rt) {
          v16bf a = load_a_frag(xs + (size_t)rt * 16 * D20c, D20c, kk * 32, lane);
          acc[c][rt] = wmma_bf16(a, b, acc[c][rt]);
        }
      }
    }
    __syncthreads();
  }
#pragma unroll
  for (int c = 0; c < CC; ++c)
#pragma unroll
    for (int rt = 0; rt < 4; ++rt)
#pragma unroll
      for (int r = 0; r < 8; ++r) {
        float hv = gelu_exact(acc[c][rt][r] + gb1[c * GK + col0 + cn]);
        ghb[((size_t)c * NN + n0 + rt * 16 + r + 8 * kh) * GK + col0 + cn] = (__bf16)hv;
      }
}

// gate out + softmax -> d_out gate region (C,N,E)
__global__ void gate2_kernel(const __bf16* __restrict__ ghb, const float* __restrict__ gW2,
                             const float* __restrict__ gb2, float* __restrict__ gwout) {
  int g = blockIdx.x * 256 + threadIdx.x;   // c*N + n
  int c = g >> 14;
  float acc[Ec];
#pragma unroll
  for (int e = 0; e < Ec; ++e) acc[e] = gb2[c * Ec + e];
  const __bf16* hv = ghb + (size_t)g * GK;
  const float* W = gW2 + (size_t)c * GK * Ec;
  for (int k = 0; k < GK; ++k) {
    float h = (float)hv[k];
#pragma unroll
    for (int e = 0; e < Ec; ++e) acc[e] += h * W[k * Ec + e];
  }
  float mx = acc[0];
#pragma unroll
  for (int e = 1; e < Ec; ++e) mx = fmaxf(mx, acc[e]);
  float s = 0.f;
#pragma unroll
  for (int e = 0; e < Ec; ++e) { acc[e] = __expf(acc[e] - mx); s += acc[e]; }
  float rs = 1.0f / s;
#pragma unroll
  for (int e = 0; e < Ec; ++e) gwout[(size_t)g * Ec + e] = acc[e] * rs;
}

// experts + gate-weighted fuse: 64-row blocks, grid (N/64, C)
__global__ void experts_kernel(const __bf16* __restrict__ xb20, const __bf16* __restrict__ xb5,
                               const __bf16* __restrict__ w1_20, const __bf16* __restrict__ w1_5,
                               const __bf16* __restrict__ w2_20, const __bf16* __restrict__ w2_5,
                               const float* __restrict__ b1_20, const float* __restrict__ b1_5,
                               const float* __restrict__ b2_20, const float* __restrict__ b2_5,
                               const float* __restrict__ gw, float* __restrict__ fused,
                               __bf16* __restrict__ fusedb) {
  extern __shared__ char smem[];
  __bf16* xs   = (__bf16*)smem;            // 64*1536 (196608 B), one scale at a time
  __bf16* hbuf = xs + 64 * D20c;           // 64*256  (32768 B)
  int tid = threadIdx.x, lane = tid & 31, w = tid >> 5;
  int n0 = blockIdx.x * 64;
  int c = blockIdx.y;
  int cn = lane & 15, kh = lane >> 4;

  v8f facc[4][2];
#pragma unroll
  for (int rt = 0; rt < 4; ++rt)
#pragma unroll
    for (int t = 0; t < 2; ++t) facc[rt][t] = vzero8();

  for (int e = 0; e < Ec; ++e) {
    int ei = e & 3;
    if (ei == 0) {   // (re)stage activations for this scale
      if (e == 4) __syncthreads();
      const __bf16* src = (e < 4) ? xb20 : xb5;
      const uint4* s = (const uint4*)(src + (size_t)n0 * D20c);
      uint4* d = (uint4*)xs;
      for (int q = tid; q < 64 * D20c / 8; q += 256) d[q] = s[q];
      __syncthreads();
    }
    const __bf16* W1 = ((e < 4) ? w1_20 : w1_5) + (size_t)(c * 4 + ei) * D20c * Hc; // [256][1536]
    const float*  b1 = ((e < 4) ? b1_20 : b1_5) + (c * 4 + ei) * Hc;
    const __bf16* W2 = ((e < 4) ? w2_20 : w2_5) + (size_t)(c * 4 + ei) * Hc * Hc;   // [256][256]
    const float*  b2 = ((e < 4) ? b2_20 : b2_5) + (c * 4 + ei) * Hc;

    // hidden 64x256: per wave 2 col-tiles x 4 row-tiles, B reused 4x
#pragma unroll
    for (int t = 0; t < 2; ++t) {
      int col0 = (w * 2 + t) * 16;
      v8f hacc[4];
#pragma unroll
      for (int rt = 0; rt < 4; ++rt) hacc[rt] = vzero8();
      for (int kk = 0; kk < D20c / 32; ++kk) {
        v16bf b = load_b_frag(W1, D20c, kk * 32, col0, lane);
#pragma unroll
        for (int rt = 0; rt < 4; ++rt) {
          v16bf a = load_a_frag(xs + (size_t)rt * 16 * D20c, D20c, kk * 32, lane);
          hacc[rt] = wmma_bf16(a, b, hacc[rt]);
        }
      }
#pragma unroll
      for (int rt = 0; rt < 4; ++rt)
#pragma unroll
        for (int r = 0; r < 8; ++r) {
          float hv = gelu_exact(hacc[rt][r] + b1[col0 + cn]);
          hbuf[(rt * 16 + r + 8 * kh) * Hc + col0 + cn] = (__bf16)hv;
        }
    }
    __syncthreads();

    float gwv[4][8];
#pragma unroll
    for (int rt = 0; rt < 4; ++rt)
#pragma unroll
      for (int r = 0; r < 8; ++r)
        gwv[rt][r] = gw[((size_t)c * NN + n0 + rt * 16 + r + 8 * kh) * Ec + e];

#pragma unroll
    for (int t = 0; t < 2; ++t) {
      int col0 = (w * 2 + t) * 16;
      v8f oacc[4];
#pragma unroll
      for (int rt = 0; rt < 4; ++rt) oacc[rt] = vzero8();
#pragma unroll
      for (int kk = 0; kk < Hc / 32; ++kk) {
        v16bf b = load_b_frag(W2, Hc, kk * 32, col0, lane);
#pragma unroll
        for (int rt = 0; rt < 4; ++rt) {
          v16bf a = load_a_frag(hbuf + (size_t)rt * 16 * Hc, Hc, kk * 32, lane);
          oacc[rt] = wmma_bf16(a, b, oacc[rt]);
        }
      }
#pragma unroll
      for (int rt = 0; rt < 4; ++rt)
#pragma unroll
        for (int r = 0; r < 8; ++r)
          facc[rt][t][r] += gwv[rt][r] * (oacc[rt][r] + b2[col0 + cn]);
    }
    __syncthreads();   // before next expert overwrites hbuf
  }

#pragma unroll
  for (int rt = 0; rt < 4; ++rt)
#pragma unroll
    for (int t = 0; t < 2; ++t) {
      int col0 = (w * 2 + t) * 16;
#pragma unroll
      for (int r = 0; r < 8; ++r) {
        size_t idx = ((size_t)c * NN + n0 + rt * 16 + r + 8 * kh) * Hc + col0 + cn;
        fused[idx] = facc[rt][t][r];
        fusedb[idx] = (__bf16)facc[rt][t][r];
      }
    }
}

// attention scorer hidden: ah = tanh(fused @ aW1 + ab1), grid (N/64, C)
__global__ void att1_kernel(const __bf16* __restrict__ fusedb, const __bf16* __restrict__ aW1T,
                            const float* __restrict__ ab1, float* __restrict__ ah) {
  extern __shared__ char smem[];
  __bf16* As = (__bf16*)smem;   // 64*256 (32768 B)
  int tid = threadIdx.x, lane = tid & 31, w = tid >> 5;
  int n0 = blockIdx.x * 64;
  int c = blockIdx.y;
  int cn = lane & 15, kh = lane >> 4;
  int col0 = w * 16;            // 128 cols / 8 waves

  {
    const uint4* s = (const uint4*)(fusedb + ((size_t)c * NN + n0) * Hc);
    uint4* d = (uint4*)As;
    for (int q = tid; q < 64 * Hc / 8; q += 256) d[q] = s[q];
  }
  __syncthreads();

  const __bf16* W = aW1T + (size_t)c * Hc * 128;   // [128][256]
  v8f acc[4];
#pragma unroll
  for (int rt = 0; rt < 4; ++rt) acc[rt] = vzero8();
#pragma unroll
  for (int kk = 0; kk < Hc / 32; ++kk) {
    v16bf b = load_b_frag(W, Hc, kk * 32, col0, lane);
#pragma unroll
    for (int rt = 0; rt < 4; ++rt) {
      v16bf a = load_a_frag(As + (size_t)rt * 16 * Hc, Hc, kk * 32, lane);
      acc[rt] = wmma_bf16(a, b, acc[rt]);
    }
  }
#pragma unroll
  for (int rt = 0; rt < 4; ++rt)
#pragma unroll
    for (int r = 0; r < 8; ++r) {
      float v = tanhf(acc[rt][r] + ab1[c * 128 + col0 + cn]);
      ah[((size_t)c * NN + n0 + rt * 16 + r + 8 * kh) * 128 + col0 + cn] = v;
    }
}

// A[c,n] = ah . aW2 + ab2 -> d_out A region
__global__ void alogits_kernel(const float* __restrict__ ah, const float* __restrict__ aW2,
                               const float* __restrict__ ab2, float* __restrict__ Aout) {
  int g = blockIdx.x * 256 + threadIdx.x;  // c*N+n
  int c = g >> 14;
  const float* v = ah + (size_t)g * 128;
  const float* wv = aW2 + c * 128;
  float acc = ab2[c];
  for (int k = 0; k < 128; ++k) acc += v[k] * wv[k];
  Aout[g] = acc;
}

// per-c softmax stats over N
__global__ void astats_kernel(const float* __restrict__ A, float* __restrict__ stats) {
  int c = blockIdx.x, tid = threadIdx.x;
  __shared__ float red[256];
  float mx = -1e30f;
  for (int n = tid; n < NN; n += 256) mx = fmaxf(mx, A[(size_t)c * NN + n]);
  red[tid] = mx; __syncthreads();
  for (int s = 128; s > 0; s >>= 1) { if (tid < s) red[tid] = fmaxf(red[tid], red[tid + s]); __syncthreads(); }
  float m = red[0]; __syncthreads();
  float se = 0.f;
  for (int n = tid; n < NN; n += 256) se += __expf(A[(size_t)c * NN + n] - m);
  red[tid] = se; __syncthreads();
  for (int s = 128; s > 0; s >>= 1) { if (tid < s) red[tid] += red[tid + s]; __syncthreads(); }
  if (tid == 0) { stats[c * 2] = m; stats[c * 2 + 1] = red[0]; }
}

// M[c,h] = sum_n softmax(A)[n] * fused[c,n,h]
__global__ void mpool_kernel(const float* __restrict__ A, const float* __restrict__ stats,
                             const float* __restrict__ fused, float* __restrict__ Mout) {
  int c = blockIdx.x, h = threadIdx.x;
  float m = stats[c * 2], rs = 1.0f / stats[c * 2 + 1];
  float acc = 0.f;
  for (int n = 0; n < NN; ++n) {
    float p = __expf(A[(size_t)c * NN + n] - m) * rs;
    acc += p * fused[((size_t)c * NN + n) * Hc + h];
  }
  Mout[c * Hc + h] = acc;
}

// tiny transformer head over (4 x 256), single workgroup of 256 threads
__global__ void head_kernel(const float* __restrict__ tWq, const float* __restrict__ tbq,
                            const float* __restrict__ tWk, const float* __restrict__ tbk,
                            const float* __restrict__ tWv, const float* __restrict__ tbv,
                            const float* __restrict__ tWo, const float* __restrict__ tbo,
                            const float* __restrict__ ln1g, const float* __restrict__ ln1b,
                            const float* __restrict__ ln2g, const float* __restrict__ ln2b,
                            const float* __restrict__ fW1, const float* __restrict__ fb1,
                            const float* __restrict__ fW2, const float* __restrict__ fb2,
                            const float* __restrict__ clW, const float* __restrict__ clb,
                            float* __restrict__ out) {
  __shared__ float Ms[CC][Hc], qs[CC][Hc], ks[CC][Hc], vs[CC][Hc];
  __shared__ float tmp[CC][Hc], ts[CC][Hc], rf[CC][Hc], h1[CC][FFc];
  __shared__ float att[8][CC][CC];
  __shared__ float mu1[CC], rs1[CC], mu2[CC], rs2[CC];
  int j = threadIdx.x;
  const float* Mmat = out + OUT_M;

  for (int c = 0; c < CC; ++c) Ms[c][j] = Mmat[c * Hc + j];
  __syncthreads();

  for (int c = 0; c < CC; ++c) {
    float aq = tbq[j], ak = tbk[j], av = tbv[j];
    for (int d = 0; d < Hc; ++d) {
      float m = Ms[c][d];
      aq += m * tWq[d * Hc + j]; ak += m * tWk[d * Hc + j]; av += m * tWv[d * Hc + j];
    }
    qs[c][j] = aq; ks[c][j] = ak; vs[c][j] = av;
  }
  __syncthreads();

  if (j < 128) {
    int hh = j >> 4, qc = (j >> 2) & 3, kc = j & 3;
    float s = 0.f;
    for (int d = 0; d < 32; ++d) s += qs[qc][hh * 32 + d] * ks[kc][hh * 32 + d];
    att[hh][qc][kc] = s * 0.17677669529663687f;   // 1/sqrt(32)
  }
  __syncthreads();
  if (j < 32) {
    int hh = j >> 2, qc = j & 3;
    float mx = -1e30f;
    for (int kc = 0; kc < CC; ++kc) mx = fmaxf(mx, att[hh][qc][kc]);
    float s = 0.f, ev[CC];
    for (int kc = 0; kc < CC; ++kc) { ev[kc] = __expf(att[hh][qc][kc] - mx); s += ev[kc]; }
    for (int kc = 0; kc < CC; ++kc) att[hh][qc][kc] = ev[kc] / s;
  }
  __syncthreads();
  {
    int hh = j >> 5;
    for (int c = 0; c < CC; ++c) {
      float s = 0.f;
      for (int kc = 0; kc < CC; ++kc) s += att[hh][c][kc] * vs[kc][j];
      tmp[c][j] = s;
    }
  }
  __syncthreads();
  for (int c = 0; c < CC; ++c) {
    float s = tbo[j];
    for (int d = 0; d < Hc; ++d) s += tmp[c][d] * tWo[d * Hc + j];
    ts[c][j] = Ms[c][j] + s;
  }
  __syncthreads();
  if (j < CC) {
    float s = 0.f, s2 = 0.f;
    for (int d = 0; d < Hc; ++d) { float v = ts[j][d]; s += v; s2 += v * v; }
    float m = s * (1.0f / Hc), var = s2 * (1.0f / Hc) - m * m;
    mu1[j] = m; rs1[j] = rsqrtf(var + 1e-5f);
  }
  __syncthreads();
  for (int c = 0; c < CC; ++c) ts[c][j] = (ts[c][j] - mu1[c]) * rs1[c] * ln1g[j] + ln1b[j];
  __syncthreads();
  if (j < FFc) {
    for (int c = 0; c < CC; ++c) {
      float s = fb1[j];
      for (int d = 0; d < Hc; ++d) s += ts[c][d] * fW1[d * FFc + j];
      h1[c][j] = fmaxf(s, 0.f);
    }
  }
  __syncthreads();
  for (int c = 0; c < CC; ++c) {
    float s = fb2[j];
    for (int f = 0; f < FFc; ++f) s += h1[c][f] * fW2[f * Hc + j];
    rf[c][j] = ts[c][j] + s;
  }
  __syncthreads();
  if (j < CC) {
    float s = 0.f, s2 = 0.f;
    for (int d = 0; d < Hc; ++d) { float v = rf[j][d]; s += v; s2 += v * v; }
    float m = s * (1.0f / Hc), var = s2 * (1.0f / Hc) - m * m;
    mu2[j] = m; rs2[j] = rsqrtf(var + 1e-5f);
  }
  __syncthreads();
  for (int c = 0; c < CC; ++c) rf[c][j] = (rf[c][j] - mu2[c]) * rs2[c] * ln2g[j] + ln2b[j];
  __syncthreads();
  for (int c = 0; c < CC; ++c) out[OUT_REFINED + c * Hc + j] = rf[c][j];
  if (j < CC) {
    int c = j;
    float s = clb[c];
    for (int d = 0; d < Hc; ++d) s += rf[c][d] * clW[c * Hc + d];
    out[OUT_LOGITS + c] = s;
    float p = 1.0f / (1.0f + __expf(-s));
    out[OUT_PROB + c] = p;
    ((int*)out)[OUT_YHAT + c] = (p > 0.5f) ? 1 : 0;
  }
}

// ---------------- launcher ----------------
extern "C" void kernel_launch(void* const* d_in, const int* in_sizes, int n_in,
                              void* d_out, int out_size, void* d_ws, size_t ws_size,
                              hipStream_t stream) {
  enum { IX = 0, IC20, IC5,
         iWp20, ibp20, igp20, ibep20, iWp5, ibp5, igp5, ibep5,
         igW1, igb1, igW2, igb2,
         ie20W1, ie20b1, ie20W2, ie20b2, ie5W1, ie5b1, ie5W2, ie5b2,
         iaW1, iab1, iaW2, iab2,
         itWq, itbq, itWk, itbk, itWv, itbv, itWo, itbo,
         iln1g, iln1b, iln2g, iln2b, ifW1, ifb1, ifW2, ifb2, iclW, iclb };
#define P(i) ((const float*)d_in[i])

  char* ws = (char*)d_ws;
  __bf16* pe20  = (__bf16*)(ws + WS_PE20);
  __bf16* pe5   = (__bf16*)(ws + WS_PE5);
  __bf16* xb20  = (__bf16*)(ws + WS_XB20);
  __bf16* xb5   = (__bf16*)(ws + WS_XB5);
  __bf16* ghb   = (__bf16*)(ws + WS_GHB);
  float*  fused = (float*) (ws + WS_FUSED);
  __bf16* fusedb= (__bf16*)(ws + WS_FUSEDB);
  float*  ah    = (float*) (ws + WS_AH);
  __bf16* y20b  = (__bf16*)(ws + WS_Y20B);
  __bf16* y5b   = (__bf16*)(ws + WS_Y5B);
  __bf16* wp20b = (__bf16*)(ws + WS_WP20B);
  __bf16* wp5b  = (__bf16*)(ws + WS_WP5B);
  __bf16* gW1b  = (__bf16*)(ws + WS_GW1B);
  __bf16* e20W1b= (__bf16*)(ws + WS_E20W1B);
  __bf16* e5W1b = (__bf16*)(ws + WS_E5W1B);
  __bf16* e20W2b= (__bf16*)(ws + WS_E20W2B);
  __bf16* e5W2b = (__bf16*)(ws + WS_E5W2B);
  __bf16* aW1b  = (__bf16*)(ws + WS_AW1B);
  float*  astats= (float*) (ws + WS_ASTATS);
  float*  out   = (float*)d_out;

  // transpose-convert weights to bf16 column-major [Ncol][K]
  auto cvtT = [&](int idx, __bf16* dst, int K, int Ncol, int batch) {
    cvt_t_kernel<<<dim3(K / 32, Ncol / 32, batch), 256, 0, stream>>>(P(idx), dst, K, Ncol);
  };
  cvtT(iWp20,  wp20b,  POSc, D20c, 1);
  cvtT(iWp5,   wp5b,   POSc, D5c, 1);
  cvtT(igW1,   gW1b,   Dtot, GK, CC);
  cvtT(ie20W1, e20W1b, D20c, Hc, 16);
  cvtT(ie5W1,  e5W1b,  D5c,  Hc, 16);
  cvtT(ie20W2, e20W2b, Hc,   Hc, 16);
  cvtT(ie5W2,  e5W2b,  Hc,   Hc, 16);
  cvtT(iaW1,   aW1b,   Hc,   GK, CC);

  posenc_kernel<<<NN * POSc / 256, 256, 0, stream>>>(P(IC20), pe20);
  posenc_kernel<<<NN * POSc / 256, 256, 0, stream>>>(P(IC5),  pe5);

  // projection GEMMs (Y bf16), then LN+residual
  proj_gemm_kernel<<<dim3(NN / 64, D20c / 256), 256, 64 * POSc * 2, stream>>>(
      pe20, wp20b, P(ibp20), y20b);
  proj_gemm_kernel<<<dim3(NN / 64, D5c / 256), 256, 64 * POSc * 2, stream>>>(
      pe5, wp5b, P(ibp5), y5b);
  size_t sm_ln = 16 * D20c * 2 + 2 * 1024 + 128;   // 51328 B
  ln_kernel<<<NN / 16, 256, sm_ln, stream>>>(y20b, P(igp20), P(ibep20), P(IX), 0, xb20);
  ln_kernel<<<NN / 16, 256, sm_ln, stream>>>(y5b, P(igp5), P(ibep5), P(IX), D20c, xb5);

  gate1_kernel<<<NN / 64, 256, (size_t)64 * D20c * 2, stream>>>(xb20, xb5, gW1b, P(igb1), ghb);
  gate2_kernel<<<CC * NN / 256, 256, 0, stream>>>(ghb, P(igW2), P(igb2), out + OUT_GATE);

  size_t sm_exp = (size_t)64 * D20c * 2 + 64 * Hc * 2;   // 229376 B
  experts_kernel<<<dim3(NN / 64, CC), 256, sm_exp, stream>>>(
      xb20, xb5, e20W1b, e5W1b, e20W2b, e5W2b,
      P(ie20b1), P(ie5b1), P(ie20b2), P(ie5b2),
      out + OUT_GATE, fused, fusedb);

  att1_kernel<<<dim3(NN / 64, CC), 256, (size_t)64 * Hc * 2, stream>>>(fusedb, aW1b, P(iab1), ah);
  alogits_kernel<<<CC * NN / 256, 256, 0, stream>>>(ah, P(iaW2), P(iab2), out + OUT_A);
  astats_kernel<<<CC, 256, 0, stream>>>(out + OUT_A, astats);
  mpool_kernel<<<CC, 256, 0, stream>>>(out + OUT_A, astats, fused, out + OUT_M);

  head_kernel<<<1, 256, 0, stream>>>(P(itWq), P(itbq), P(itWk), P(itbk), P(itWv), P(itbv),
                                     P(itWo), P(itbo), P(iln1g), P(iln1b), P(iln2g), P(iln2b),
                                     P(ifW1), P(ifb1), P(ifW2), P(ifb2), P(iclW), P(iclb), out);
#undef P
}
